// Contrastive_model_19061064860121
// MI455X (gfx1250) — matrix operand order repaired and checked by reading, not compiled, not run on hardware
//
#include <hip/hip_runtime.h>
#include <hip/hip_bf16.h>
#include <math.h>

#define BS 8192
#define DIN 1024
#define DEMB 256
#define NLAB 10
#define EPSN 1e-8f

typedef _Float16 f16;
typedef __attribute__((ext_vector_type(16))) _Float16 v16h;
typedef __attribute__((ext_vector_type(8)))  float    v8f;

union Frag {
  v16h h;
  unsigned u[8];
  uint4 q[2];
};

__device__ __forceinline__ v8f wmma16(v16h a, v16h b, v8f c) {
  return __builtin_amdgcn_wmma_f32_16x16x32_f16(false, a, false, b, (short)0, c,
                                                false, false);
}

__device__ __forceinline__ float waveRedSum(float x) {
  #pragma unroll
  for (int off = 16; off > 0; off >>= 1) x += __shfl_xor(x, off, 32);
  return x;
}

// ---------------- workspace layout (bytes) ----------------
#define OFF_EN16 0                        // BS*DEMB f16 = 4 MB
#define OFF_NEG  (4u*1024u*1024u)         // BS f32
#define OFF_P    (OFF_NEG  + BS*4u)       // BS f32
#define OFF_N1   (OFF_P    + BS*4u)       // BS f32
#define OFF_LN   (OFF_N1   + BS*4u)       // NLAB*DEMB f32
#define OFF_COLT (OFF_LN   + NLAB*DEMB*4u)// 16 f32
#define OFF_OWN  (OFF_COLT + 64u)         // 16 f32
#define OFF_CNT  (OFF_OWN  + 64u)         // 16 i32
#define OFF_ACC  (OFF_CNT  + 64u)         // 1 f64

// ---------------- init + label histogram ----------------
__global__ __launch_bounds__(256) void k_init(float* neg_sum, float* col_tot,
                                              float* own_col, int* counts,
                                              double* acc) {
  int i = blockIdx.x * 256 + threadIdx.x;
  if (i < BS) neg_sum[i] = 0.f;
  if (i < 16) { col_tot[i] = 0.f; own_col[i] = 0.f; counts[i] = 0; }
  if (i == 0) acc[0] = 0.0;
}

__global__ __launch_bounds__(256) void k_hist(const int* __restrict__ labels,
                                              int* counts) {
  int i = blockIdx.x * 256 + threadIdx.x;
  if (i < BS) atomicAdd(&counts[labels[i]], 1);
}

// ---------------- e = emb @ W^T + b, normalize, emit f16 ----------------
__global__ __launch_bounds__(256)
void k_transform(const float* __restrict__ emb, const float* __restrict__ W,
                 const float* __restrict__ bias, f16* __restrict__ en) {
  __shared__ __align__(16) f16 lds_a[16 * 32];
  __shared__ float lds_e[16 * 256];
  __shared__ float lds_inv[16];
  const int tid = threadIdx.x;
  const int wave = tid >> 5, lane = tid & 31;
  const int col = lane & 15, hi = lane >> 4;
  const int i0 = blockIdx.x * 16;
  const int n0 = wave * 32;

  int koffB[8];
  #pragma unroll
  for (int v = 0; v < 8; ++v) koffB[v] = 2 * v + 16 * hi;

  v8f c0 = {}, c1 = {};
  for (int kk = 0; kk < DIN; kk += 32) {
    __syncthreads();
    {
      int row = tid >> 4, cp = (tid & 15) * 2;
      const float* src = emb + (size_t)(i0 + row) * DIN + kk + cp;
      lds_a[row * 32 + cp]     = (f16)src[0];
      lds_a[row * 32 + cp + 1] = (f16)src[1];
    }
    __syncthreads();
    Frag a, b0, b1;
    // A fragment: two contiguous 16B chunks per lane -> ds_load_b128
    a.q[0] = *(const uint4*)&lds_a[col * 32 + 16 * hi];
    a.q[1] = *(const uint4*)&lds_a[col * 32 + 16 * hi + 8];
    #pragma unroll
    for (int v = 0; v < 8; ++v) {
      const float* w0 = W + (size_t)(n0 + col) * DIN + kk + koffB[v];
      const float* w1 = W + (size_t)(n0 + 16 + col) * DIN + kk + koffB[v];
      b0.h[2 * v] = (f16)w0[0]; b0.h[2 * v + 1] = (f16)w0[1];
      b1.h[2 * v] = (f16)w1[0]; b1.h[2 * v + 1] = (f16)w1[1];
    }
    c0 = wmma16(a.h, b0.h, c0);
    c1 = wmma16(a.h, b1.h, c1);
  }
  float bv0 = bias[n0 + col], bv1 = bias[n0 + 16 + col];
  #pragma unroll
  for (int v = 0; v < 8; ++v) {
    int m = v + 8 * hi;
    lds_e[m * 256 + n0 + col]      = c0[v] + bv0;
    lds_e[m * 256 + n0 + 16 + col] = c1[v] + bv1;
  }
  __syncthreads();
  for (int r = wave * 2; r < wave * 2 + 2; ++r) {
    float s = 0.f;
    for (int cc = lane; cc < 256; cc += 32) { float x = lds_e[r * 256 + cc]; s += x * x; }
    s = waveRedSum(s);
    if (lane == 0) lds_inv[r] = 1.0f / fmaxf(sqrtf(s), EPSN);
  }
  __syncthreads();
  for (int idx = tid; idx < 16 * 256; idx += 256) {
    int r = idx >> 8;
    en[(size_t)(i0 + r) * DEMB + (idx & 255)] = (f16)(lds_e[idx] * lds_inv[r]);
  }
}

// helper: load a B fragment (en row j, k-chunk kk) as two b128 loads
__device__ __forceinline__ Frag loadBFrag(const f16* brow, int kk, int hi) {
  Frag b;
  const uint4* p = (const uint4*)(brow + kk * 32 + 16 * hi);
  b.q[0] = p[0];
  b.q[1] = p[1];
  return b;
}

// helper: load an A fragment from staged LDS rows
__device__ __forceinline__ Frag loadAFrag(const f16* lrow, int kk, int hi) {
  Frag a;
  a.q[0] = *(const uint4*)(lrow + kk * 32 + 16 * hi);
  a.q[1] = *(const uint4*)(lrow + kk * 32 + 16 * hi + 8);
  return a;
}

// ---------------- pass 1 over C: neg_sum_i = sum_{lab_j != lab_i} exp(C_ij) ----
__global__ __launch_bounds__(256)
void k_negsum(const f16* __restrict__ en, const int* __restrict__ labels,
              float* __restrict__ neg_sum) {
  __shared__ __align__(16) f16 lds_i[32 * 256];
  __shared__ int lab_i[32];
  __shared__ float neg_lds[32];
  const int tid = threadIdx.x;
  const int wave = tid >> 5, lane = tid & 31;
  const int col = lane & 15, hi = lane >> 4;
  const int i0 = blockIdx.x * 32;
  const int jtile0 = blockIdx.y * 128;

  {
    const uint4* src = (const uint4*)(en + (size_t)i0 * DEMB);
    uint4* dst = (uint4*)lds_i;
    for (int idx = tid; idx < 32 * 256 / 8; idx += 256) dst[idx] = src[idx];
    if (tid < 32) { lab_i[tid] = labels[i0 + tid]; neg_lds[tid] = 0.f; }
  }
  __syncthreads();

  Frag a0[8], a1[8];
  #pragma unroll
  for (int kk = 0; kk < 8; ++kk) {
    a0[kk] = loadAFrag(&lds_i[col * 256], kk, hi);
    a1[kk] = loadAFrag(&lds_i[(16 + col) * 256], kk, hi);
  }
  int li0[8], li1[8];
  #pragma unroll
  for (int v = 0; v < 8; ++v) { li0[v] = lab_i[v + 8 * hi]; li1[v] = lab_i[16 + v + 8 * hi]; }

  float na0[8] = {0,0,0,0,0,0,0,0}, na1[8] = {0,0,0,0,0,0,0,0};
  for (int t = 0; t < 16; ++t) {
    const int j0 = (jtile0 + wave + 8 * t) * 16;
    const int lj = labels[j0 + col];
    const f16* brow = en + (size_t)(j0 + col) * DEMB;
    v8f c0 = {}, c1 = {};
    #pragma unroll
    for (int kk = 0; kk < 8; ++kk) {
      Frag b = loadBFrag(brow, kk, hi);
      c0 = wmma16(a0[kk].h, b.h, c0);
      c1 = wmma16(a1[kk].h, b.h, c1);
    }
    #pragma unroll
    for (int v = 0; v < 8; ++v) {
      if (lj != li0[v]) na0[v] += __expf(c0[v]);
      if (lj != li1[v]) na1[v] += __expf(c1[v]);
    }
  }
  #pragma unroll
  for (int v = 0; v < 8; ++v) {
    atomicAdd(&neg_lds[v + 8 * hi], na0[v]);
    atomicAdd(&neg_lds[16 + v + 8 * hi], na1[v]);
  }
  __syncthreads();
  if (tid < 32) atomicAdd(&neg_sum[i0 + tid], neg_lds[tid]);
}

// ---- pass 2 over C: sum_{same,i!=j} (-C + log(neg_sum_i + exp(C))) ----
__global__ __launch_bounds__(256)
void k_inter(const f16* __restrict__ en, const int* __restrict__ labels,
             const float* __restrict__ neg_sum, double* __restrict__ acc_same) {
  __shared__ __align__(16) f16 lds_i[32 * 256];
  __shared__ int lab_i[32];
  __shared__ float ns_lds[32];
  __shared__ float blockacc;
  const int tid = threadIdx.x;
  const int wave = tid >> 5, lane = tid & 31;
  const int col = lane & 15, hi = lane >> 4;
  const int i0 = blockIdx.x * 32;
  const int jtile0 = blockIdx.y * 128;

  {
    const uint4* src = (const uint4*)(en + (size_t)i0 * DEMB);
    uint4* dst = (uint4*)lds_i;
    for (int idx = tid; idx < 32 * 256 / 8; idx += 256) dst[idx] = src[idx];
    if (tid < 32) { lab_i[tid] = labels[i0 + tid]; ns_lds[tid] = neg_sum[i0 + tid]; }
    if (tid == 0) blockacc = 0.f;
  }
  __syncthreads();

  Frag a0[8], a1[8];
  #pragma unroll
  for (int kk = 0; kk < 8; ++kk) {
    a0[kk] = loadAFrag(&lds_i[col * 256], kk, hi);
    a1[kk] = loadAFrag(&lds_i[(16 + col) * 256], kk, hi);
  }
  int li0[8], li1[8]; float ns0[8], ns1[8];
  #pragma unroll
  for (int v = 0; v < 8; ++v) {
    li0[v] = lab_i[v + 8 * hi];      ns0[v] = ns_lds[v + 8 * hi];
    li1[v] = lab_i[16 + v + 8 * hi]; ns1[v] = ns_lds[16 + v + 8 * hi];
  }

  float term = 0.f;
  for (int t = 0; t < 16; ++t) {
    const int j0 = (jtile0 + wave + 8 * t) * 16;
    const int jg = j0 + col;
    const int lj = labels[jg];
    const f16* brow = en + (size_t)jg * DEMB;
    v8f c0 = {}, c1 = {};
    #pragma unroll
    for (int kk = 0; kk < 8; ++kk) {
      Frag b = loadBFrag(brow, kk, hi);
      c0 = wmma16(a0[kk].h, b.h, c0);
      c1 = wmma16(a1[kk].h, b.h, c1);
    }
    #pragma unroll
    for (int v = 0; v < 8; ++v) {
      int ig0 = i0 + v + 8 * hi;
      int ig1 = i0 + 16 + v + 8 * hi;
      if (lj == li0[v] && ig0 != jg) { float c = c0[v]; term += -c + logf(ns0[v] + __expf(c)); }
      if (lj == li1[v] && ig1 != jg) { float c = c1[v]; term += -c + logf(ns1[v] + __expf(c)); }
    }
  }
  term = waveRedSum(term);
  if (lane == 0) atomicAdd(&blockacc, term);
  __syncthreads();
  if (tid == 0) atomicAdd(acc_same, (double)blockacc);
}

// ---------------- normalize label prototypes ----------------
__global__ __launch_bounds__(256)
void k_normproto(const float* __restrict__ lemb, float* __restrict__ ln) {
  __shared__ float red[8];
  __shared__ float inv;
  const int r = blockIdx.x, t = threadIdx.x;
  float x = lemb[r * DEMB + t];
  float s = waveRedSum(x * x);
  if ((t & 31) == 0) red[t >> 5] = s;
  __syncthreads();
  if (t == 0) {
    float tot = 0.f;
    for (int w = 0; w < 8; ++w) tot += red[w];
    inv = 1.0f / fmaxf(sqrtf(tot), EPSN);
  }
  __syncthreads();
  ln[r * DEMB + t] = x * inv;
}

// ---------------- S = en @ ln^T -> P, neg1, column sums ----------------
__global__ __launch_bounds__(256)
void k_proto(const f16* __restrict__ en, const float* __restrict__ ln,
             const int* __restrict__ labels, float* __restrict__ P,
             float* __restrict__ neg1, float* col_tot, float* own_col) {
  __shared__ float col_part[NLAB], own_part[NLAB];
  const int tid = threadIdx.x, wave = tid >> 5, lane = tid & 31;
  if (tid < NLAB) { col_part[tid] = 0.f; own_part[tid] = 0.f; }
  __syncthreads();
  const int i = blockIdx.x * 8 + wave;
  const f16* er = en + (size_t)i * DEMB;
  float ev[8];
  #pragma unroll
  for (int t = 0; t < 8; ++t) ev[t] = (float)er[lane * 8 + t];
  float S[NLAB];
  #pragma unroll
  for (int l = 0; l < NLAB; ++l) {
    const float* lr = ln + l * DEMB;
    float s = 0.f;
    #pragma unroll
    for (int t = 0; t < 8; ++t) s += ev[t] * lr[lane * 8 + t];
    S[l] = waveRedSum(s);
  }
  if (lane == 0) {
    int lab = labels[i];
    float n1 = 0.f;
    #pragma unroll
    for (int l = 0; l < NLAB; ++l) {
      float E = __expf(S[l]);
      if (l != lab) n1 += E; else atomicAdd(&own_part[l], E);
      atomicAdd(&col_part[l], E);
    }
    P[i] = S[lab];
    neg1[i] = n1;
  }
  __syncthreads();
  if (tid < NLAB) {
    atomicAdd(&col_tot[tid], col_part[tid]);
    atomicAdd(&own_col[tid], own_part[tid]);
  }
}

// ---------------- final scalar assembly (fp64) ----------------
__global__ __launch_bounds__(256)
void k_final(const float* __restrict__ P, const float* __restrict__ neg1,
             const float* __restrict__ neg_sum, const int* __restrict__ labels,
             const float* __restrict__ col_tot, const float* __restrict__ own_col,
             const int* __restrict__ counts, const double* __restrict__ acc_same,
             float* __restrict__ out) {
  __shared__ double r1[256], r2[256], r3[256];
  double s1 = 0.0, s2 = 0.0, sd = 0.0;
  for (int i = threadIdx.x; i < BS; i += 256) {
    float Pv = P[i];
    float eP = __expf(Pv);
    int lab = labels[i];
    s1 += (double)(-Pv + logf(neg1[i] + eP));
    float n2 = col_tot[lab] - own_col[lab];
    s2 += (double)(-Pv + logf(n2 + eP));
    sd += (double)(BS - counts[lab]) * (double)logf(neg_sum[i] + 1.0f);
  }
  r1[threadIdx.x] = s1; r2[threadIdx.x] = s2; r3[threadIdx.x] = sd;
  __syncthreads();
  if (threadIdx.x == 0) {
    double S1 = 0.0, S2 = 0.0, SD = 0.0;
    for (int k = 0; k < 256; ++k) { S1 += r1[k]; S2 += r2[k]; SD += r3[k]; }
    double inter = (acc_same[0] + SD) / ((double)BS * (double)BS);
    double l1 = S1 / (double)BS, l2 = S2 / (double)BS;
    out[0] = (float)(0.5 * inter + 0.5 * (l1 + l2));
  }
}

extern "C" void kernel_launch(void* const* d_in, const int* in_sizes, int n_in,
                              void* d_out, int out_size, void* d_ws, size_t ws_size,
                              hipStream_t stream) {
  (void)in_sizes; (void)n_in; (void)out_size; (void)ws_size;
  const float* emb   = (const float*)d_in[0];
  const int*   labels= (const int*)d_in[1];
  const float* W     = (const float*)d_in[2];
  const float* bias  = (const float*)d_in[3];
  const float* lemb  = (const float*)d_in[4];
  float* out = (float*)d_out;

  char* ws = (char*)d_ws;
  f16*    en      = (f16*)(ws + OFF_EN16);
  float*  neg_sum = (float*)(ws + OFF_NEG);
  float*  P       = (float*)(ws + OFF_P);
  float*  neg1    = (float*)(ws + OFF_N1);
  float*  ln      = (float*)(ws + OFF_LN);
  float*  col_tot = (float*)(ws + OFF_COLT);
  float*  own_col = (float*)(ws + OFF_OWN);
  int*    counts  = (int*)(ws + OFF_CNT);
  double* acc     = (double*)(ws + OFF_ACC);

  k_init<<<(BS + 255) / 256, 256, 0, stream>>>(neg_sum, col_tot, own_col, counts, acc);
  k_hist<<<(BS + 255) / 256, 256, 0, stream>>>(labels, counts);
  k_transform<<<BS / 16, 256, 0, stream>>>(emb, W, bias, en);
  k_normproto<<<NLAB, 256, 0, stream>>>(lemb, ln);
  k_negsum<<<dim3(BS / 32, 4), 256, 0, stream>>>(en, labels, neg_sum);
  k_proto<<<BS / 8, 256, 0, stream>>>(en, ln, labels, P, neg1, col_tot, own_col);
  k_inter<<<dim3(BS / 32, 4), 256, 0, stream>>>(en, labels, neg_sum, acc);
  k_final<<<1, 256, 0, stream>>>(P, neg1, neg_sum, labels, col_tot, own_col, counts, acc, out);
}
